// CustomSelfAttention_39573828665783
// MI455X (gfx1250) — matrix operand order repaired and checked by reading, not compiled, not run on hardware
//
#include <hip/hip_runtime.h>

typedef __bf16 v16bf __attribute__((ext_vector_type(16)));
typedef float  v8f   __attribute__((ext_vector_type(8)));

#define BB 4
#define SS 2048
#define DD 1024
#define HH 16
#define HD 64

static __device__ inline v8f wmma_bf16(v16bf a, v16bf b, v8f c) {
  return __builtin_amdgcn_wmma_f32_16x16x32_bf16(false, a, false, b, (short)0, c, false, false);
}

// Low 32 bits of a flat LDS pointer = LDS byte offset (ISA 10.2 aperture rules).
static __device__ inline unsigned lds_addr(const void* p) {
  return (unsigned)(unsigned long long)p;
}

// CDNA5 async global->LDS 128b copy (tracked by ASYNCcnt). VDST = LDS byte addr.
static __device__ inline void async_load_b128(unsigned lds_off, const void* gaddr) {
  asm volatile("global_load_async_to_lds_b128 %0, %1, off"
               :: "v"(lds_off), "v"(gaddr) : "memory");
}

#if __has_builtin(__builtin_amdgcn_s_wait_asynccnt)
#define WAIT_ASYNC(n) __builtin_amdgcn_s_wait_asynccnt(n)
#else
#define WAIT_ASYNC(n) asm volatile("s_wait_asynccnt %0" :: "i"(n) : "memory")
#endif

static __device__ inline void wait_ds0() {
  asm volatile("s_wait_dscnt 0x0" ::: "memory");
}

// A fragment: 16x32 bf16 from row-major LDS tile p[row][ld].
// Lane holds row m = lane%16; k = (lane/16)*8 + {0..7, 16..23}  (ISA 7.12.2)
static __device__ inline v16bf load_a_frag(const __bf16* p, int ld, int row0, int k0, int lane) {
  const int m = lane & 15;
  const int koff = (lane >> 4) << 3;
  const __bf16* base = p + (row0 + m) * ld + k0 + koff;
  v16bf a;
#pragma unroll
  for (int i = 0; i < 16; ++i) a[i] = base[((i & 8) << 1) + (i & 7)];
  return a;
}

// B fragment built from a K-TRANSPOSED LDS tile p[n][ld] (contraction dim contiguous):
// lane holds column n = lane%16; k = (lane/16)*8 + {0..7, 16..23}
static __device__ inline v16bf load_bT_frag(const __bf16* p, int ld, int k0, int n0, int lane) {
  const int n = lane & 15;
  const int kb = (lane >> 4) << 3;
  const __bf16* base = p + (n0 + n) * ld + k0 + kb;
  v16bf b;
#pragma unroll
  for (int i = 0; i < 16; ++i) b[i] = base[((i & 8) << 1) + (i & 7)];
  return b;
}

// ---------------- QKV projection: C_bf16[8192x1024] = x_f32 @ W_f32 + b ----------------
// V output is stored head-transposed: vbt[b][h][hd][s]
__global__ __launch_bounds__(256) void qkv_kernel(
    const float* __restrict__ x,
    const float* __restrict__ Wq, const float* __restrict__ bq,
    const float* __restrict__ Wk, const float* __restrict__ bk,
    const float* __restrict__ Wv, const float* __restrict__ bv,
    __bf16* __restrict__ qb, __bf16* __restrict__ kb, __bf16* __restrict__ vbt) {
  const float* W; const float* bias; __bf16* out;
  if (blockIdx.z == 0)      { W = Wq; bias = bq; out = qb; }
  else if (blockIdx.z == 1) { W = Wk; bias = bk; out = kb; }
  else                      { W = Wv; bias = bv; out = vbt; }
  const bool vtr = (blockIdx.z == 2);

  __shared__ alignas(16) __bf16 sA[128 * 32];   // [row][k]
  __shared__ alignas(16) __bf16 sWt[64 * 32];   // [n][k]  (transposed)

  const int tid = threadIdx.x;
  const int lane = tid & 31;
  const int w = tid >> 5;
  const int m0 = blockIdx.x * 128;
  const int n0 = blockIdx.y * 64;

  v8f acc[4] = {};

  for (int kk = 0; kk < DD; kk += 32) {
    __syncthreads();
#pragma unroll
    for (int it = 0; it < 4; ++it) {                    // A: 128x32 f32 -> bf16
      const int t = tid + it * 256;
      const int r = t >> 3, seg = t & 7;
      const float4 f = ((const float4*)(x + (size_t)(m0 + r) * DD + kk))[seg];
      union { __bf16 h[4]; uint2 u; } pk;
      pk.h[0] = (__bf16)f.x; pk.h[1] = (__bf16)f.y;
      pk.h[2] = (__bf16)f.z; pk.h[3] = (__bf16)f.w;
      ((uint2*)(sA + r * 32 + seg * 4))[0] = pk.u;
    }
#pragma unroll
    for (int it = 0; it < 2; ++it) {                    // W: column reads -> sWt[n][k]
      const int t = tid + it * 256;
      const int n = t & 63, kq = t >> 6;                // coalesced across lanes per k-row
      const float* wp = W + (size_t)(kk + kq * 4) * DD + n0 + n;
      union { __bf16 h[4]; uint2 u; } pk;
      pk.h[0] = (__bf16)wp[0];
      pk.h[1] = (__bf16)wp[DD];
      pk.h[2] = (__bf16)wp[2 * DD];
      pk.h[3] = (__bf16)wp[3 * DD];
      ((uint2*)(sWt + n * 32 + kq * 4))[0] = pk.u;
    }
    __syncthreads();
    v16bf a = load_a_frag(sA, 32, w * 16, 0, lane);
#pragma unroll
    for (int t = 0; t < 4; ++t)
      acc[t] = wmma_bf16(a, load_bT_frag(sWt, 32, 0, t * 16, lane), acc[t]);
  }

  const int half8 = (lane >> 4) << 3;
  const int n = lane & 15;
#pragma unroll
  for (int t = 0; t < 4; ++t) {
    const int c = n0 + t * 16 + n;
    const float bb = bias[c];
#pragma unroll
    for (int j = 0; j < 8; ++j) {
      const int r = m0 + w * 16 + half8 + j;
      const __bf16 val = (__bf16)(acc[t][j] + bb);
      if (vtr) {                                        // V: [b][h][hd][s]
        const int bidx = r >> 11, s = r & (SS - 1);
        const int h = c >> 6, hd = c & 63;
        out[(((size_t)bidx * HH + h) * HD + hd) * SS + s] = val;
      } else {
        out[(size_t)r * DD + c] = val;
      }
    }
  }
}

// ---------------- Flash attention: per-(b,h), 128 queries / block ----------------
// Double-buffered async K/V^T pipeline; buffer 1 reuses the Q-tile LDS region.
__global__ __launch_bounds__(256) void attn_kernel(
    const __bf16* __restrict__ qbuf, const __bf16* __restrict__ kbuf,
    const __bf16* __restrict__ vbt, const float* __restrict__ amm,
    const unsigned char* __restrict__ kpm, __bf16* __restrict__ ob) {
  const int q0 = blockIdx.x * 128;
  const int b  = blockIdx.y >> 4;
  const int h  = blockIdx.y & 15;

  // carve: [0..8191] Q tile (later K1|V1), [8192..16383] K0|V0, [16384..24575] P staging
  __shared__ alignas(16) __bf16 smem[24576];            // 48 KB
  __bf16* sQ  = smem;
  __bf16* myP = smem + 16384 + (threadIdx.x >> 5) * 1024;

  const int tid = threadIdx.x;
  const int lane = tid & 31;
  const int w = tid >> 5;
  const int half8 = (lane >> 4) << 3;
  const int nlane = lane & 15;

  const size_t headoff = (size_t)b * SS * DD + (size_t)h * HD;        // q/k rows
  const size_t vhead   = ((size_t)b * HH + h) * HD * SS;              // v^T rows

#pragma unroll
  for (int it = 0; it < 4; ++it) {                      // Q tile via async DMA (4 ops/wave)
    const int t = tid + it * 256;
    const int r = t >> 3, seg = t & 7;
    async_load_b128(lds_addr(sQ) + t * 16,
                    qbuf + headoff + (size_t)(q0 + r) * DD + seg * 8);
  }
#pragma unroll
  for (int it = 0; it < 2; ++it) {                      // chunk 0 -> buffer 0 (4 ops/wave)
    const int t = tid + it * 256;
    const int r = t >> 3, seg = t & 7;
    async_load_b128(lds_addr(smem + 8192) + t * 16,
                    kbuf + headoff + (size_t)r * DD + seg * 8);
    async_load_b128(lds_addr(smem + 12288) + t * 16,
                    vbt + vhead + (size_t)r * SS + seg * 8);
  }
  WAIT_ASYNC(4);                                        // in-order: Q's 4 ops retired
  __syncthreads();

  v16bf qf0 = load_a_frag(sQ, 64, w * 16, 0, lane);
  v16bf qf1 = load_a_frag(sQ, 64, w * 16, 32, lane);
  wait_ds0();                                           // Q reads done before region reuse

  float mrow[8], lrow[8];
  v8f o[4] = {};
  bool qmask[8];
  const float* ammrow[8];
#pragma unroll
  for (int j = 0; j < 8; ++j) {
    mrow[j] = -1e30f; lrow[j] = 0.0f;
    const int qr = q0 + w * 16 + half8 + j;
    qmask[j]  = kpm[(size_t)b * SS + qr] != 0;          // query-axis padding mask
    ammrow[j] = amm + ((size_t)b * SS + qr) * SS;
  }

  for (int i = 0; i < 32; ++i) {
    const int k0 = i * 64;
    const int odd = i & 1;
    const __bf16* sK  = odd ? smem         : smem + 8192;   // runtime selects (no
    const __bf16* sVt = odd ? smem + 4096  : smem + 12288;  //  static LDS-pointer init)

    WAIT_ASYNC(0);                                      // chunk i landed in buf[i&1]
    __syncthreads();                                    // all waves done with buf[(i+1)&1]

    if (i < 31) {                                       // issue chunk i+1 under compute i
      __bf16* nK = odd ? smem + 8192  : smem;
      __bf16* nV = odd ? smem + 12288 : smem + 4096;
#pragma unroll
      for (int it = 0; it < 2; ++it) {
        const int t = tid + it * 256;
        const int r = t >> 3, seg = t & 7;
        async_load_b128(lds_addr(nK) + t * 16,
                        kbuf + headoff + (size_t)(k0 + 64 + r) * DD + seg * 8);
        async_load_b128(lds_addr(nV) + t * 16,
                        vbt + vhead + (size_t)r * SS + k0 + 64 + seg * 8);
      }
#pragma unroll
      for (int j = 0; j < 8; ++j)                       // prefetch next modifier rows
        __builtin_prefetch(ammrow[j] + k0 + 64 + nlane, 0, 0);
    }

    v8f e[4];
#pragma unroll
    for (int t = 0; t < 4; ++t) {                       // E = Q . K^T (contract hd=64)
      v8f z = {};
      z    = wmma_bf16(qf0, load_bT_frag(sK, 64, 0,  t * 16, lane), z);
      e[t] = wmma_bf16(qf1, load_bT_frag(sK, 64, 32, t * 16, lane), z);
    }

    float ev[4][8];
#pragma unroll
    for (int t = 0; t < 4; ++t)
#pragma unroll
      for (int j = 0; j < 8; ++j) {
        const float mod = ammrow[j][k0 + t * 16 + nlane];
        ev[t][j] = qmask[j] ? -1e10f : e[t][j] * 0.125f * mod;
      }

    float p[4][8], alpha[8];
#pragma unroll
    for (int j = 0; j < 8; ++j) {                       // online softmax per row
      float rm = fmaxf(fmaxf(ev[0][j], ev[1][j]), fmaxf(ev[2][j], ev[3][j]));
      rm = fmaxf(rm, __shfl_xor(rm, 1, 32));
      rm = fmaxf(rm, __shfl_xor(rm, 2, 32));
      rm = fmaxf(rm, __shfl_xor(rm, 4, 32));
      rm = fmaxf(rm, __shfl_xor(rm, 8, 32));
      const float mnew = fmaxf(mrow[j], rm);
      alpha[j] = __expf(mrow[j] - mnew);
      float rs = 0.0f;
#pragma unroll
      for (int t = 0; t < 4; ++t) { p[t][j] = __expf(ev[t][j] - mnew); rs += p[t][j]; }
      rs += __shfl_xor(rs, 1, 32);
      rs += __shfl_xor(rs, 2, 32);
      rs += __shfl_xor(rs, 4, 32);
      rs += __shfl_xor(rs, 8, 32);
      lrow[j] = lrow[j] * alpha[j] + rs;
      mrow[j] = mnew;
    }
#pragma unroll
    for (int t = 0; t < 4; ++t)
#pragma unroll
      for (int j = 0; j < 8; ++j) o[t][j] *= alpha[j];

#pragma unroll
    for (int t = 0; t < 4; ++t)                         // C-layout -> A-layout via LDS
#pragma unroll
      for (int j = 0; j < 8; ++j)
        myP[(half8 + j) * 64 + t * 16 + nlane] = (__bf16)p[t][j];
    v16bf pa0 = load_a_frag(myP, 64, 0, 0, lane);
    v16bf pa1 = load_a_frag(myP, 64, 0, 32, lane);

#pragma unroll
    for (int t = 0; t < 4; ++t) {                       // O += P . V (contract keys=64)
      o[t] = wmma_bf16(pa0, load_bT_frag(sVt, 64, 0,  t * 16, lane), o[t]);
      o[t] = wmma_bf16(pa1, load_bT_frag(sVt, 64, 32, t * 16, lane), o[t]);
    }
  }

#pragma unroll
  for (int j = 0; j < 8; ++j) {
    const float inv = 1.0f / lrow[j];
    const size_t row = (size_t)b * SS + (q0 + w * 16 + half8 + j);
#pragma unroll
    for (int t = 0; t < 4; ++t)
      ob[row * DD + h * HD + t * 16 + nlane] = (__bf16)(o[t][j] * inv);
  }
}

// ---------------- Output projection: out_f32 = A_bf16 @ Wo + bo ----------------
__global__ __launch_bounds__(256) void oproj_kernel(
    const __bf16* __restrict__ A, const float* __restrict__ W,
    const float* __restrict__ bias, float* __restrict__ out) {
  __shared__ alignas(16) __bf16 sA[128 * 32];
  __shared__ alignas(16) __bf16 sWt[64 * 32];           // [n][k] transposed

  const int tid = threadIdx.x;
  const int lane = tid & 31;
  const int w = tid >> 5;
  const int m0 = blockIdx.x * 128;
  const int n0 = blockIdx.y * 64;

  v8f acc[4] = {};

  for (int kk = 0; kk < DD; kk += 32) {
    __syncthreads();
#pragma unroll
    for (int it = 0; it < 2; ++it) {                    // A: 128x32 bf16 copy
      const int t = tid + it * 256;
      const int r = t >> 2, seg = t & 3;
      ((uint4*)(sA + r * 32))[seg] =
          ((const uint4*)(A + (size_t)(m0 + r) * DD + kk))[seg];
    }
#pragma unroll
    for (int it = 0; it < 2; ++it) {                    // W: column reads -> sWt[n][k]
      const int t = tid + it * 256;
      const int n = t & 63, kq = t >> 6;
      const float* wp = W + (size_t)(kk + kq * 4) * DD + n0 + n;
      union { __bf16 h[4]; uint2 u; } pk;
      pk.h[0] = (__bf16)wp[0];
      pk.h[1] = (__bf16)wp[DD];
      pk.h[2] = (__bf16)wp[2 * DD];
      pk.h[3] = (__bf16)wp[3 * DD];
      ((uint2*)(sWt + n * 32 + kq * 4))[0] = pk.u;
    }
    __syncthreads();
    v16bf a = load_a_frag(sA, 32, w * 16, 0, lane);
#pragma unroll
    for (int t = 0; t < 4; ++t)
      acc[t] = wmma_bf16(a, load_bT_frag(sWt, 32, 0, t * 16, lane), acc[t]);
  }

  const int half8 = (lane >> 4) << 3;
  const int n = lane & 15;
#pragma unroll
  for (int t = 0; t < 4; ++t) {
    const int c = n0 + t * 16 + n;
    const float bb = bias[c];
#pragma unroll
    for (int j = 0; j < 8; ++j)
      out[(size_t)(m0 + w * 16 + half8 + j) * DD + c] = acc[t][j] + bb;
  }
}

extern "C" void kernel_launch(void* const* d_in, const int* in_sizes, int n_in,
                              void* d_out, int out_size, void* d_ws, size_t ws_size,
                              hipStream_t stream) {
  const float* x          = (const float*)d_in[0];
  const unsigned char* km = (const unsigned char*)d_in[1];
  const float* amm        = (const float*)d_in[2];
  const float* Wq = (const float*)d_in[3];
  const float* bq = (const float*)d_in[4];
  const float* Wk = (const float*)d_in[5];
  const float* bk = (const float*)d_in[6];
  const float* Wv = (const float*)d_in[7];
  const float* bv = (const float*)d_in[8];
  const float* Wo = (const float*)d_in[9];
  const float* bo = (const float*)d_in[10];
  float* out = (float*)d_out;

  const size_t NE = (size_t)BB * SS * DD;               // 8,388,608 elements
  char* wsb = (char*)d_ws;
  __bf16* qb  = (__bf16*)(wsb);
  __bf16* kb  = (__bf16*)(wsb + 2 * NE);
  __bf16* vbt = (__bf16*)(wsb + 4 * NE);                // head-transposed V
  __bf16* ab  = (__bf16*)(wsb + 6 * NE);                // 64 MB total

  dim3 blk(256);
  hipLaunchKernelGGL(qkv_kernel, dim3((BB * SS) / 128, DD / 64, 3), blk, 0, stream,
                     x, Wq, bq, Wk, bk, Wv, bv, qb, kb, vbt);
  hipLaunchKernelGGL(attn_kernel, dim3(SS / 128, BB * HH), blk, 0, stream,
                     qb, kb, vbt, amm, km, ab);
  hipLaunchKernelGGL(oproj_kernel, dim3((BB * SS) / 128, DD / 64), blk, 0, stream,
                     ab, Wo, bo, out);
}
